// APGATLayer_22239340659364
// MI455X (gfx1250) — compile-verified, matrix operand order checked by reading, not run on hardware
//
#include <hip/hip_runtime.h>
#include <hip/hip_bf16.h>

typedef __attribute__((ext_vector_type(16))) __bf16        v16bf;
typedef __attribute__((ext_vector_type(8)))  float         v8f;
typedef __attribute__((ext_vector_type(8)))  unsigned short us8;

#define H_DIM   512
#define F_DIM   128
#define HD      256
#define NHEAD   8
#define ODIM    32

__device__ __forceinline__ unsigned short f32_to_bf16(float f) {
    unsigned int u = __float_as_uint(f);
    unsigned int r = (u + 0x7FFFu + ((u >> 16) & 1u)) >> 16;
    return (unsigned short)r;
}

__device__ __forceinline__ void atomicMaxFloat(float* addr, float val) {
    // int-punning max: works for mixed signs given a negative init sentinel
    if (val >= 0.0f) atomicMax((int*)addr, __float_as_int(val));
    else             atomicMin((unsigned int*)addr, __float_as_uint(val));
}

// ---------------- init: zero out, reset segment max/denom ----------------
__global__ void init_kernel(float* __restrict__ out, int out_n,
                            float* __restrict__ segmax, float* __restrict__ denom, int segn) {
    int t = blockIdx.x * blockDim.x + threadIdx.x;
    if (t < out_n) out[t] = 0.0f;
    if (t < segn) { segmax[t] = -3.0e38f; denom[t] = 0.0f; }
}

// ---------------- convert W_fc (256x512 f32) -> bf16 ----------------
__global__ void cvt_wfc_kernel(const float* __restrict__ W_fc, unsigned short* __restrict__ wbf) {
    int t = blockIdx.x * blockDim.x + threadIdx.x;
    if (t < HD * H_DIM) wbf[t] = f32_to_bf16(W_fc[t]);
}

// ------- V[h][k] = sum_d W_feat[h*32+d][k] * a_feat[d]  (a_feat = W_attn[64..95]) -------
__global__ void v_kernel(const float* __restrict__ W_feat, const float* __restrict__ W_attn,
                         float* __restrict__ V) {
    int t = blockIdx.x * blockDim.x + threadIdx.x;
    if (t >= NHEAD * F_DIM) return;
    int hh = t >> 7, k = t & (F_DIM - 1);
    float s = 0.0f;
#pragma unroll
    for (int d = 0; d < ODIM; ++d)
        s += W_feat[(hh * ODIM + d) * F_DIM + k] * W_attn[2 * ODIM + d];
    V[t] = s;
}

// ---------------- z = h @ W_fc^T via WMMA bf16 (M=Na, N=256, K=512) ----------------
// Block: 256 threads (8 waves), tile BM=32 x BN=256, BK=32, 16 K-steps.
__global__ __launch_bounds__(256) void gemm_z_kernel(const float* __restrict__ hmat,
                                                     const unsigned short* __restrict__ wbf,
                                                     float* __restrict__ z, int Na) {
    // fragments staged in WMMA lane order: [tile][lane][elem]
    __shared__ __align__(16) unsigned short AsF[2][32][16];   // 2 M-tiles
    __shared__ __align__(16) unsigned short BsF[16][32][16];  // 16 N-tiles

    const int tid  = threadIdx.x;
    const int wave = tid >> 5;
    const int lane = tid & 31;
    const int m0   = blockIdx.x * 32;
    const int mt   = wave >> 2;        // this wave's M tile (0/1)
    const int nb   = (wave & 3) * 4;   // this wave's first N tile (owns 4)

    v8f acc0 = {}, acc1 = {}, acc2 = {}, acc3 = {};

    for (int kk = 0; kk < H_DIM / 32; ++kk) {
        // ---- stage A: 32 rows x 32 k of h, f32 -> bf16, into fragment order ----
        {
            int r  = tid >> 3;           // 0..31 local row
            int kb = (tid & 7) * 4;      // k base, float4 per thread
            float4 v = make_float4(0.f, 0.f, 0.f, 0.f);
            int row = m0 + r;
            if (row < Na)
                v = *(const float4*)(hmat + (long)row * H_DIM + kk * 32 + kb);
            float fv[4] = {v.x, v.y, v.z, v.w};
#pragma unroll
            for (int i = 0; i < 4; ++i) {
                int k    = kb + i;
                int half = (k & 15) >> 3;
                int e    = (k < 16) ? (k & 7) : (8 + (k & 7));
                int ln   = (r & 15) + (half << 4);
                AsF[r >> 4][ln][e] = f32_to_bf16(fv[i]);
            }
        }
        // ---- stage B: thread t owns W_fc row n=t, 32 bf16 (4x b128 loads) ----
        {
            const unsigned short* p = wbf + (long)tid * H_DIM + kk * 32;
            us8 d0 = *(const us8*)(p);
            us8 d1 = *(const us8*)(p + 8);
            us8 d2 = *(const us8*)(p + 16);
            us8 d3 = *(const us8*)(p + 24);
            unsigned short tmp[32];
#pragma unroll
            for (int i = 0; i < 8; ++i) {
                tmp[i] = d0[i]; tmp[8 + i] = d1[i]; tmp[16 + i] = d2[i]; tmp[24 + i] = d3[i];
            }
#pragma unroll
            for (int k = 0; k < 32; ++k) {
                int half = (k & 15) >> 3;
                int e    = (k < 16) ? (k & 7) : (8 + (k & 7));
                int ln   = (tid & 15) + (half << 4);
                BsF[tid >> 4][ln][e] = tmp[k];
            }
        }
        __syncthreads();

        v16bf a = *(const v16bf*)(&AsF[mt][lane][0]);
        v16bf b0 = *(const v16bf*)(&BsF[nb + 0][lane][0]);
        v16bf b1 = *(const v16bf*)(&BsF[nb + 1][lane][0]);
        v16bf b2 = *(const v16bf*)(&BsF[nb + 2][lane][0]);
        v16bf b3 = *(const v16bf*)(&BsF[nb + 3][lane][0]);
        acc0 = __builtin_amdgcn_wmma_f32_16x16x32_bf16(false, a, false, b0, (short)0, acc0, false, false);
        acc1 = __builtin_amdgcn_wmma_f32_16x16x32_bf16(false, a, false, b1, (short)0, acc1, false, false);
        acc2 = __builtin_amdgcn_wmma_f32_16x16x32_bf16(false, a, false, b2, (short)0, acc2, false, false);
        acc3 = __builtin_amdgcn_wmma_f32_16x16x32_bf16(false, a, false, b3, (short)0, acc3, false, false);
        __syncthreads();
    }

    // ---- epilogue: C/D layout — VGPR v: M = v + 8*(lane/16), N = lane%16 ----
    const int half = lane >> 4;
    const int n0   = lane & 15;
    v8f accs[4] = {acc0, acc1, acc2, acc3};
#pragma unroll
    for (int j = 0; j < 4; ++j) {
        int ng = (nb + j) * 16 + n0;
#pragma unroll
        for (int v = 0; v < 8; ++v) {
            int m = m0 + mt * 16 + v + 8 * half;
            if (m < Na) z[(long)m * HD + ng] = accs[j][v];
        }
    }
}

// ---------------- s_src[n,h] = z[n, h*32 .. ] . a_src ----------------
__global__ void s_src_kernel(const float* __restrict__ z, const float* __restrict__ W_attn,
                             float* __restrict__ s_src, int Na) {
    __shared__ float a[ODIM];
    if (threadIdx.x < ODIM) a[threadIdx.x] = W_attn[threadIdx.x];
    __syncthreads();
    int t = blockIdx.x * blockDim.x + threadIdx.x;
    if (t >= Na * NHEAD) return;
    int n = t >> 3, hh = t & 7;
    const float* zp = z + (long)n * HD + hh * ODIM;
    float s = 0.0f;
#pragma unroll
    for (int d = 0; d < ODIM; ++d) s += zp[d] * a[d];
    s_src[t] = s;
}

// ---------------- s_feat[e,h] = srl_emb[e,:] . V[h,:] (wave per edge) ----------------
__global__ __launch_bounds__(256) void s_feat_kernel(const float* __restrict__ srl,
                                                     const float* __restrict__ V,
                                                     float* __restrict__ s_feat, int E) {
    __shared__ float Vs[NHEAD][F_DIM];
    for (int i = threadIdx.x; i < NHEAD * F_DIM; i += blockDim.x)
        Vs[i >> 7][i & (F_DIM - 1)] = V[i];
    __syncthreads();
    int wave   = blockIdx.x * (blockDim.x >> 5) + (threadIdx.x >> 5);
    int lane   = threadIdx.x & 31;
    int nwaves = gridDim.x * (blockDim.x >> 5);
    for (int e = wave; e < E; e += nwaves) {
        const float* p = srl + (long)e * F_DIM;
        float x0 = p[lane], x1 = p[lane + 32], x2 = p[lane + 64], x3 = p[lane + 96];
        float part[NHEAD];
#pragma unroll
        for (int hh = 0; hh < NHEAD; ++hh)
            part[hh] = x0 * Vs[hh][lane] + x1 * Vs[hh][lane + 32]
                     + x2 * Vs[hh][lane + 64] + x3 * Vs[hh][lane + 96];
#pragma unroll
        for (int hh = 0; hh < NHEAD; ++hh) {
            float s = part[hh];
#pragma unroll
            for (int off = 16; off > 0; off >>= 1) s += __shfl_xor(s, off, 32);
            part[hh] = s;
        }
        if (lane < NHEAD) s_feat[(long)e * NHEAD + lane] = part[lane];
    }
}

// ---------------- leaky_relu + segment max ----------------
__global__ void edge_max_kernel(const int* __restrict__ src, const int* __restrict__ dst,
                                const float* __restrict__ s_src, const float* __restrict__ s_feat,
                                float* __restrict__ e_buf, float* __restrict__ segmax, int E) {
    int t = blockIdx.x * blockDim.x + threadIdx.x;
    if (t >= E * NHEAD) return;
    int e = t >> 3, hh = t & 7;
    float v  = s_src[src[e] * NHEAD + hh] + s_feat[t];
    float lr = v > 0.0f ? v : 0.01f * v;
    e_buf[t] = lr;
    atomicMaxFloat(&segmax[dst[e] * NHEAD + hh], lr);
}

// ---------------- exp(e - max) + segment sum ----------------
__global__ void edge_exp_kernel(const int* __restrict__ dst, float* __restrict__ e_buf,
                                const float* __restrict__ segmax, float* __restrict__ denom, int E) {
    int t = blockIdx.x * blockDim.x + threadIdx.x;
    if (t >= E * NHEAD) return;
    int e = t >> 3, hh = t & 7;
    float m  = segmax[dst[e] * NHEAD + hh];
    float ex = __expf(e_buf[t] - m);
    e_buf[t] = ex;
    atomicAdd(&denom[dst[e] * NHEAD + hh], ex);
}

// ---------------- out[dst] += alpha * z[src]  (block per edge, 256 lanes = 256 cols) ----------------
__global__ __launch_bounds__(256) void scatter_kernel(const int* __restrict__ src, const int* __restrict__ dst,
                                                      const float* __restrict__ e_buf,
                                                      const float* __restrict__ denom,
                                                      const float* __restrict__ z,
                                                      float* __restrict__ out, int E) {
    int e = blockIdx.x;
    int c = threadIdx.x;
    int hh = c >> 5;
    int s = src[e], d = dst[e];
    float dn    = denom[d * NHEAD + hh];
    float alpha = e_buf[(long)e * NHEAD + hh] / (dn > 0.0f ? dn : 1.0f);
    atomicAdd(&out[(long)d * HD + c], alpha * z[(long)s * HD + c]);
}

extern "C" void kernel_launch(void* const* d_in, const int* in_sizes, int n_in,
                              void* d_out, int out_size, void* d_ws, size_t ws_size,
                              hipStream_t stream) {
    const float* hmat   = (const float*)d_in[0];
    const float* srl    = (const float*)d_in[1];
    const int*   src    = (const int*)d_in[2];
    const int*   dst    = (const int*)d_in[3];
    const float* W_fc   = (const float*)d_in[4];
    const float* W_feat = (const float*)d_in[5];
    const float* W_attn = (const float*)d_in[6];
    float*       out    = (float*)d_out;

    const int Na = in_sizes[0] / H_DIM;
    const int E  = in_sizes[2];
    const int Np = out_size / HD;

    // workspace layout (floats)
    float* ws = (float*)d_ws;
    size_t off = 0;
    float*          z      = ws + off; off += (size_t)Na * HD;
    unsigned short* wbf    = (unsigned short*)(ws + off); off += (HD * H_DIM) / 2;
    float*          V      = ws + off; off += NHEAD * F_DIM;
    float*          s_src  = ws + off; off += (size_t)Na * NHEAD;
    float*          s_feat = ws + off; off += (size_t)E * NHEAD;
    float*          e_buf  = ws + off; off += (size_t)E * NHEAD;
    float*          segmax = ws + off; off += (size_t)Np * NHEAD;
    float*          denom  = ws + off; off += (size_t)Np * NHEAD;

    int initN = out_size > Np * NHEAD ? out_size : Np * NHEAD;
    init_kernel<<<(initN + 255) / 256, 256, 0, stream>>>(out, out_size, segmax, denom, Np * NHEAD);
    cvt_wfc_kernel<<<(HD * H_DIM + 255) / 256, 256, 0, stream>>>(W_fc, wbf);
    v_kernel<<<(NHEAD * F_DIM + 255) / 256, 256, 0, stream>>>(W_feat, W_attn, V);
    gemm_z_kernel<<<(Na + 31) / 32, 256, 0, stream>>>(hmat, wbf, z, Na);
    s_src_kernel<<<(Na * NHEAD + 255) / 256, 256, 0, stream>>>(z, W_attn, s_src, Na);
    s_feat_kernel<<<4096, 256, 0, stream>>>(srl, V, s_feat, E);
    edge_max_kernel<<<(E * NHEAD + 255) / 256, 256, 0, stream>>>(src, dst, s_src, s_feat, e_buf, segmax, E);
    edge_exp_kernel<<<(E * NHEAD + 255) / 256, 256, 0, stream>>>(dst, e_buf, segmax, denom, E);
    scatter_kernel<<<E, 256, 0, stream>>>(src, dst, e_buf, denom, z, out, E);
}
